// ContextAwareModule_52201032516134
// MI455X (gfx1250) — matrix operand order repaired and checked by reading, not compiled, not run on hardware
//
#include <hip/hip_runtime.h>
#include <cstdint>

typedef __attribute__((ext_vector_type(16))) __bf16 v16bf;
typedef __attribute__((ext_vector_type(8)))  float  v8f;

#define HH 48
#define WW 48
#define NPIX 2304      // 48*48
#define NC   2116      // 46*46
#define NCP  2176      // padded to mult of 128 (17 tiles of 128)
#define KP   576       // 9*64
#define NP2  640       // padded 576 -> 5 tiles of 128

// ---------------- prep: bg = g*mask, per-pixel sum g^2, sum bg^2 -------------
__global__ __launch_bounds__(256)
void prep_kernel(const float* __restrict__ g, const float* __restrict__ mask,
                 float* __restrict__ bg, float* __restrict__ sqg,
                 float* __restrict__ sqbg)
{
    int t   = threadIdx.x;
    int pix = blockIdx.x * 4 + (t >> 6);
    int ch  = t & 63;
    float gv = g[pix * 64 + ch];
    float mk = mask[pix];
    float bv = gv * mk;
    bg[pix * 64 + ch] = bv;
    __shared__ float r1[256], r2[256];
    r1[t] = gv * gv;
    r2[t] = bv * bv;
    __syncthreads();
    for (int s = 32; s > 0; s >>= 1) {
        if (ch < s) { r1[t] += r1[t + s]; r2[t] += r2[t + s]; }
        __syncthreads();
    }
    if (ch == 0) { sqg[pix] = r1[t]; sqbg[pix] = r2[t]; }
}

// ---------------- 3x3 neighborhood sums: wwd (same-pad), k1d (valid) ---------
__global__ void sums3x3_kernel(const float* __restrict__ sqg,
                               const float* __restrict__ sqbg,
                               float* __restrict__ wwd, float* __restrict__ k1d)
{
    int i = blockIdx.x * blockDim.x + threadIdx.x;
    if (i < NPIX) {
        int y = i / WW, x = i % WW;
        float s = 0.f;
        for (int dy = -1; dy <= 1; ++dy)
            for (int dx = -1; dx <= 1; ++dx) {
                int yy = y + dy, xx = x + dx;
                if (yy >= 0 && yy < HH && xx >= 0 && xx < WW)
                    s += sqg[yy * WW + xx];
            }
        wwd[i] = s;
    } else if (i < NPIX + NCP) {
        int j = i - NPIX;
        float s = 0.f;
        if (j < NC) {
            int jy = j / 46, jx = j % 46;
            for (int dy = 0; dy < 3; ++dy)
                for (int dx = 0; dx < 3; ++dx)
                    s += sqbg[(jy + dy) * WW + (jx + dx)];
        }
        k1d[j] = s;   // padded cols -> 0
    }
}

// ---------------- helper: p1 patch value -------------------------------------
__device__ __forceinline__ float p1_val(const float* __restrict__ bg, int j, int k)
{
    int dy = k / 192, dx = (k / 64) % 3, ch = k & 63;
    int jy = j / 46, jx = j % 46;
    return bg[((jy + dy) * WW + (jx + dx)) * 64 + ch];
}

// ---------------- A operand (P2) plain row-major bf16 ------------------------
__global__ void build_p2_kernel(const float* __restrict__ g, __bf16* __restrict__ p2)
{
    int idx = blockIdx.x * blockDim.x + threadIdx.x;
    if (idx >= NPIX * KP) return;
    int m = idx / KP, k = idx % KP;
    int dy = k / 192, dx = (k / 64) % 3, ch = k & 63;
    int y = m / WW - 1 + dy, x = m % WW - 1 + dx;
    float v = (y >= 0 && y < HH && x >= 0 && x < WW) ? g[(y * WW + x) * 64 + ch] : 0.f;
    p2[idx] = (__bf16)v;
}

// --- B operands packed in WMMA-fragment order: ------------------------------
// Bp[((kc*Ntot + n)*2 + hi)*16 + e] = Blogical[kc*32 + 16*hi + e][n]
// GEMM1 B: Blogical[k][j] = p1[j][k]   (Ktot=576,  Ntot=2176)
__global__ void build_bp1_kernel(const float* __restrict__ bg, __bf16* __restrict__ bp)
{
    int idx = blockIdx.x * blockDim.x + threadIdx.x;
    if (idx >= KP * NCP) return;
    int e = idx & 15, block = idx >> 4;
    int hi = block & 1, rest = block >> 1;
    int j = rest % NCP, kc = rest / NCP;
    int k = kc * 32 + 16 * hi + e;
    float v = (j < NC) ? p1_val(bg, j, k) : 0.f;
    bp[idx] = (__bf16)v;
}

// GEMM2 B: Blogical[j][n] = p1[j][n]   (Ktot=2176, Ntot=640)
__global__ void build_bp2_kernel(const float* __restrict__ bg, __bf16* __restrict__ bp)
{
    int idx = blockIdx.x * blockDim.x + threadIdx.x;
    if (idx >= NCP * NP2) return;
    int e = idx & 15, block = idx >> 4;
    int hi = block & 1, rest = block >> 1;
    int n = rest % NP2, kc = rest / NP2;
    int j = kc * 32 + 16 * hi + e;
    float v = (j < NC && n < KP) ? p1_val(bg, j, n) : 0.f;
    bp[idx] = (__bf16)v;
}

// ---------------- tiled bf16 WMMA GEMM, 128x128 macro-tile ------------------
// double-buffered LDS + register-pipelined B fragments, fused epilogue
// mode 0: C = rowTerm[m] + colTerm[n] - 2*(A@B)   (DS1 from CS)
// mode 1: C = rowTerm[m] * (A@B)                  (softmax-normalized T)
#define BSTRIDE 12   // u32 per 16-elem block in LDS (48B, 16B-aligned)
__global__ __launch_bounds__(256)
void gemm_wmma_kernel(const uint32_t* __restrict__ A, int lda,   // u32 stride
                      const uint32_t* __restrict__ Bp, int Ntot, // packed-B, logical N
                      float* __restrict__ Cmat, int ldc,
                      int kChunks, int mode,
                      const float* __restrict__ rowTerm,
                      const float* __restrict__ colTerm)
{
    __shared__ uint32_t sA[2][128 * 16];        // 128 rows x 32 bf16, double buffered
    __shared__ uint32_t sB[2][256 * BSTRIDE];   // 256 fragment-blocks, double buffered
    const int t    = threadIdx.x;
    const int wv   = t >> 5;
    const int lane = t & 31;
    const int hi   = lane >> 4;
    const int lo   = lane & 15;
    const int m0   = blockIdx.x * 128;
    const int n0   = blockIdx.y * 128;

    const int aRow = t >> 1, aHalf = (t & 1) * 8;
    const uint32_t* aSrcBase = A + (size_t)(m0 + aRow) * lda + aHalf;
    const uint32_t* bSrcBase = Bp + (size_t)n0 * 16 + t * 8;
    const int aDst = aRow * 16 + aHalf;
    const int bDst = t * BSTRIDE;

    v8f acc[8] = {};
    uint4 a0, a1, b0, b1;

    // preload chunk 0 into buffer 0
    a0 = *(const uint4*)(aSrcBase);
    a1 = *(const uint4*)(aSrcBase + 4);
    b0 = *(const uint4*)(bSrcBase);
    b1 = *(const uint4*)(bSrcBase + 4);
    { uint4* dA = (uint4*)&sA[0][aDst]; dA[0] = a0; dA[1] = a1; }
    { uint4* dB = (uint4*)&sB[0][bDst]; dB[0] = b0; dB[1] = b1; }
    __syncthreads();

    const int mrow = wv * 16 + lo;
    for (int kc = 0; kc < kChunks; ++kc) {
        const int cur = kc & 1, nxt = cur ^ 1;
        const bool more = (kc + 1 < kChunks);
        if (more) {   // issue next chunk's global loads early (latency hidden by WMMAs)
            const uint32_t* sa = aSrcBase + (kc + 1) * 16;
            const uint32_t* sb = bSrcBase + (size_t)(kc + 1) * Ntot * 16;
            a0 = *(const uint4*)(sa);
            a1 = *(const uint4*)(sa + 4);
            b0 = *(const uint4*)(sb);
            b1 = *(const uint4*)(sb + 4);
            if (kc + 2 < kChunks) {   // global_prefetch_b8, distance 2
                __builtin_prefetch(aSrcBase + (kc + 2) * 16, 0, 3);
                __builtin_prefetch(bSrcBase + (size_t)(kc + 2) * Ntot * 16, 0, 3);
            }
        }

        // A fragment: M = wv*16 + lo ; K(e) = e + 16*hi  -> 2x ds_load_b128
        union { v16bf v; uint32_t u[8]; } af;
#pragma unroll
        for (int i = 0; i < 8; ++i) {
            int e = 2 * i;
            int k = e + 16 * hi;
            af.u[i] = sA[cur][mrow * 16 + (k >> 1)];
        }

        // B fragments: register double-buffer, load nt+1 while WMMA nt runs
        union bfrag { v16bf v; uint4 q[2]; } bf[2];
        const uint32_t* sBc = sB[cur];
        {
            const uint32_t* bb = &sBc[(lo * 2 + hi) * BSTRIDE];
            bf[0].q[0] = *(const uint4*)(bb);
            bf[0].q[1] = *(const uint4*)(bb + 4);
        }
#pragma unroll
        for (int nt = 0; nt < 8; ++nt) {
            if (nt < 7) {
                const uint32_t* bb = &sBc[(((nt + 1) * 16 + lo) * 2 + hi) * BSTRIDE];
                bf[(nt + 1) & 1].q[0] = *(const uint4*)(bb);
                bf[(nt + 1) & 1].q[1] = *(const uint4*)(bb + 4);
            }
            acc[nt] = __builtin_amdgcn_wmma_f32_16x16x32_bf16(
                false, af.v, false, bf[nt & 1].v, (short)0, acc[nt], false, false);
        }

        if (more) {   // stash next chunk into the other LDS buffer
            uint4* dA = (uint4*)&sA[nxt][aDst]; dA[0] = a0; dA[1] = a1;
            uint4* dB = (uint4*)&sB[nxt][bDst]; dB[0] = b0; dB[1] = b1;
        }
        __syncthreads();
    }

    // epilogue: C layout  M = r + 8*hi, N = lo  per 16x16 subtile
#pragma unroll
    for (int nt = 0; nt < 8; ++nt) {
        int n = n0 + nt * 16 + lo;
#pragma unroll
        for (int r = 0; r < 8; ++r) {
            int m = m0 + wv * 16 + r + 8 * hi;
            float v = acc[nt][r];
            float o;
            if (mode == 0) o = rowTerm[m] + colTerm[n] - 2.f * v;
            else           o = v * rowTerm[m];
            Cmat[(size_t)m * ldc + n] = o;
        }
    }
}

// ---------------- row mean/std -> -tanh -> exp(50*...), store bf16 E ---------
__global__ __launch_bounds__(256)
void row_softmax_kernel(const float* __restrict__ DS1, __bf16* __restrict__ E,
                        float* __restrict__ scale)
{
    int m = blockIdx.x;
    const float* row = DS1 + (size_t)m * NCP;
    __bf16* erow = E + (size_t)m * NCP;
    __shared__ float red[256];
    int t = threadIdx.x;

    float s = 0.f, ss = 0.f;
    for (int j = t; j < NC; j += 256) { float v = row[j]; s += v; ss += v * v; }
    red[t] = s; __syncthreads();
    for (int st = 128; st > 0; st >>= 1) { if (t < st) red[t] += red[t + st]; __syncthreads(); }
    float mu = red[0] * (1.f / NC);
    __syncthreads();
    red[t] = ss; __syncthreads();
    for (int st = 128; st > 0; st >>= 1) { if (t < st) red[t] += red[t + st]; __syncthreads(); }
    float var = red[0] * (1.f / NC) - mu * mu;
    __syncthreads();
    float sd  = sqrtf(fmaxf(var, 0.f));
    float rsd = 1.f / fmaxf(sd, 1e-20f);

    float se = 0.f;
    for (int j = t; j < NCP; j += 256) {
        float e = 0.f;
        if (j < NC) {
            float z = (row[j] - mu) * rsd;
            e = expf(-50.f * tanhf(z));
        }
        erow[j] = (__bf16)e;     // padded cols -> 0 (safe as GEMM2 A-padding)
        se += e;
    }
    red[t] = se; __syncthreads();
    for (int st = 128; st > 0; st >>= 1) { if (t < st) red[t] += red[t + st]; __syncthreads(); }
    if (t == 0) scale[m] = 1.f / (9.f * red[0]);
}

// ---------------- final: 9-shift combine, mask blend, 128x64 matvec, ELU -----
__global__ __launch_bounds__(256)
void final_kernel(const float* __restrict__ g, const float* __restrict__ mask,
                  const float* __restrict__ bg, const float* __restrict__ T,
                  const float* __restrict__ W2, const float* __restrict__ b2,
                  float* __restrict__ out)
{
    __shared__ float con[4][128];
    int t   = threadIdx.x;
    int lp  = t >> 6;
    int d   = t & 63;
    int pix = blockIdx.x * 4 + lp;
    int y = pix / WW, x = pix % WW;

    float acl = 0.f;
#pragma unroll
    for (int dy = 0; dy < 3; ++dy)
#pragma unroll
        for (int dx = 0; dx < 3; ++dx) {
            int yy = y + 1 - dy, xx = x + 1 - dx;
            if (yy >= 0 && yy < HH && xx >= 0 && xx < WW)
                acl += T[(size_t)(yy * WW + xx) * NP2 + (dy * 3 + dx) * 64 + d];
        }
    float mk  = mask[pix];
    float gv  = g[pix * 64 + d];
    float ACL = bg[pix * 64 + d] + acl * (1.f - mk);   // acl already has /(9*sum)
    con[lp][d]      = gv;
    con[lp][64 + d] = ACL;
    __syncthreads();

    float a = b2[d];
#pragma unroll 4
    for (int k = 0; k < 128; ++k) a += con[lp][k] * W2[k * 64 + d];
    out[pix * 64 + d] = (a > 0.f) ? a : (expf(a) - 1.f);
}

// ------------------------------- launcher ------------------------------------
extern "C" void kernel_launch(void* const* d_in, const int* in_sizes, int n_in,
                              void* d_out, int out_size, void* d_ws, size_t ws_size,
                              hipStream_t stream)
{
    const float* g_in = (const float*)d_in[0];
    const float* mask = (const float*)d_in[1];
    const float* W2   = (const float*)d_in[2];
    const float* b2   = (const float*)d_in[3];
    float* out        = (float*)d_out;

    char* p = (char*)d_ws;
    auto take = [&](size_t bytes) -> char* {
        char* r = p;
        p += (bytes + 255) & ~(size_t)255;
        return r;
    };
    float*  bg    = (float*)take((size_t)NPIX * 64 * 4);
    float*  sqg   = (float*)take((size_t)NPIX * 4);
    float*  sqbg  = (float*)take((size_t)NPIX * 4);
    float*  wwd   = (float*)take((size_t)NPIX * 4);
    float*  k1d   = (float*)take((size_t)NCP * 4);
    float*  scale = (float*)take((size_t)NPIX * 4);
    __bf16* p2b   = (__bf16*)take((size_t)NPIX * KP * 2);
    __bf16* bp1   = (__bf16*)take((size_t)KP * NCP * 2);    // packed GEMM1 B
    __bf16* bp2   = (__bf16*)take((size_t)NCP * NP2 * 2);   // packed GEMM2 B
    float*  DS1   = (float*)take((size_t)NPIX * NCP * 4);
    __bf16* Eb    = (__bf16*)take((size_t)NPIX * NCP * 2);
    float*  T     = (float*)take((size_t)NPIX * NP2 * 4);

    for (int b = 0; b < 4; ++b) {
        const float* g  = g_in + (size_t)b * NPIX * 64;
        const float* mk = mask + (size_t)b * NPIX;
        float*       ob = out  + (size_t)b * NPIX * 64;

        prep_kernel<<<NPIX / 4, 256, 0, stream>>>(g, mk, bg, sqg, sqbg);
        build_p2_kernel <<<(NPIX * KP) / 256, 256, 0, stream>>>(g, p2b);
        build_bp1_kernel<<<(KP * NCP)  / 256, 256, 0, stream>>>(bg, bp1);
        build_bp2_kernel<<<(NCP * NP2) / 256, 256, 0, stream>>>(bg, bp2);
        sums3x3_kernel  <<<(NPIX + NCP + 255) / 256, 256, 0, stream>>>(sqg, sqbg, wwd, k1d);

        // GEMM1: DS1 = k1d[n] + wwd[m] - 2 * (P2 @ P1^T)   M=2304 N=2176 K=576
        gemm_wmma_kernel<<<dim3(NPIX / 128, NCP / 128), 256, 0, stream>>>(
            (const uint32_t*)p2b, KP / 2, (const uint32_t*)bp1, NCP,
            DS1, NCP, KP / 32, 0, wwd, k1d);

        row_softmax_kernel<<<NPIX, 256, 0, stream>>>(DS1, Eb, scale);

        // GEMM2: T = scale[m] * (E @ P1)   M=2304 N=640 K=2176
        gemm_wmma_kernel<<<dim3(NPIX / 128, NP2 / 128), 256, 0, stream>>>(
            (const uint32_t*)Eb, NCP / 2, (const uint32_t*)bp2, NP2,
            T, NP2, NCP / 32, 1, scale, nullptr);

        final_kernel<<<NPIX / 4, 256, 0, stream>>>(g, mk, bg, T, W2, b2, ob);
    }
}
